// VectorQuantizer_1168231104757
// MI455X (gfx1250) — matrix operand order repaired and checked by reading, not compiled, not run on hardware
//
#include <hip/hip_runtime.h>
#include <stdint.h>

// ---------------- problem constants ----------------
#define BB   16
#define DD   64
#define HH   64
#define WW   64
#define KK   1024
#define NTOT (BB*HH*WW)          // 65536 query rows
#define QELEMS (BB*DD*HH*WW)     // 4194304 quantized output elems
// d_out layout: [0, QELEMS) quantized, QELEMS dict_loss, QELEMS+1 commit_loss,
//               [QELEMS+2, QELEMS+2+NTOT) indices (as float)
#define IDX_BASE (QELEMS + 2)

#define KC      64               // codes staged per LDS chunk
#define NCHUNK  (KK / KC)        // 16
#define CHUNK_H (KC * DD)        // 4096 ushorts per (hi|lo) plane
#define CHUNK_U (2 * CHUNK_H)    // 8192 ushorts = 16 KB per packed chunk
#define WAVES   8
#define TPB     (WAVES * 32)

typedef __bf16         v16bf __attribute__((ext_vector_type(16)));
typedef float          v8f   __attribute__((ext_vector_type(8)));
typedef unsigned short us16  __attribute__((ext_vector_type(16)));
typedef unsigned int   v4u   __attribute__((ext_vector_type(4)));
typedef int            v4i   __attribute__((ext_vector_type(4)));
typedef int            v8i   __attribute__((ext_vector_type(8)));

#if defined(__has_builtin)
#  if __has_builtin(__builtin_amdgcn_tensor_load_to_lds)
#    define HAVE_TDM 1
#  else
#    define HAVE_TDM 0
#  endif
#else
#  define HAVE_TDM 0
#endif

__device__ __forceinline__ unsigned short f2bf(float f) {
  unsigned u = __float_as_uint(f);
  unsigned r = (u + 0x7FFFu + ((u >> 16) & 1u)) >> 16;   // RNE
  return (unsigned short)r;
}
__device__ __forceinline__ float bf2f(unsigned short h) {
  return __uint_as_float(((unsigned)h) << 16);
}

#if HAVE_TDM
// One-shot TDM DMA of a packed 16 KB chunk (4096 dwords) global -> LDS.
// D# group0: count=1, lds_addr, 57-bit global_addr, type=2 ("image").
// D# group1: data_size=4B, tensor_dim0=4096, tensor_dim1=1, tile_dim0=4096,
//            tile_dim1=1, tensor_dim0_stride=4096.  Groups 2/3: zero (<=2D).
__device__ __forceinline__ void tdm_load_chunk(const unsigned short* gsrc,
                                               unsigned short* ldst) {
  unsigned long long ga = (unsigned long long)(uintptr_t)gsrc;
  unsigned ldsOff = (unsigned)(uintptr_t)ldst;      // LDS aperture: addr[31:0] = LDS byte offset
  v4u g0 = { 1u,                                     // count=1 (valid), user mode
             ldsOff,
             (unsigned)ga,
             ((unsigned)(ga >> 32) & 0x01FFFFFFu) | 0x80000000u };   // global_addr[56:32] | type=2
  v8i g1 = { (int)(2u << 16),          // workgroup_mask=0, data_size=2 (4 bytes)
             (int)(4096u << 16),       // tensor_dim0[15:0] @ bits 63:48
             (int)0x00010000,          // tensor_dim0[31:16]=0, tensor_dim1=1
             (int)(4096u << 16),       // tile_dim0=4096 @ bits 127:112
             1,                        // tile_dim1=1, tile_dim2=0
             4096,                     // tensor_dim0_stride[31:0]
             0, 0 };
  v4i z4 = { 0, 0, 0, 0 };
#if __clang_major__ >= 23
  v8i z8 = { 0, 0, 0, 0, 0, 0, 0, 0 };
  __builtin_amdgcn_tensor_load_to_lds(g0, g1, z4, z4, z8, 0);
#else
  __builtin_amdgcn_tensor_load_to_lds(g0, g1, z4, z4, 0);
#endif
}
#endif

// ---------------- kernel 0: prep codebook ----------------
// E [D=64][K=1024] f32  ->  Epk: per 64-code chunk, [hi 8KB | lo 8KB] bf16,
// code-major within chunk (B-fragment-ready), + e_norm2 [K] f32.
__global__ __launch_bounds__(256) void vq_prep(const float* __restrict__ E,
                                               unsigned short* __restrict__ Epk,
                                               float* __restrict__ En2) {
  int k = blockIdx.x * 256 + threadIdx.x;
  if (k >= KK) return;
  const int c    = k >> 6;           // chunk
  const int kl   = k & 63;           // code within chunk
  const int base = c * CHUNK_U + kl * DD;
  float s = 0.f;
  #pragma unroll 8
  for (int d = 0; d < DD; ++d) {
    float v = E[d * KK + k];                 // coalesced across lanes (k contiguous)
    s += v * v;
    unsigned short h = f2bf(v);
    unsigned short l = f2bf(v - bf2f(h));
    Epk[base + d]           = h;
    Epk[base + CHUNK_H + d] = l;
  }
  En2[k] = s;
}

// ---------------- kernel 1: fused GEMM + argmin + gather + loss partials ----
__global__ __launch_bounds__(TPB) void vq_main(const float* __restrict__ X,
                                               const float* __restrict__ E,
                                               const unsigned short* __restrict__ Epk,
                                               const float* __restrict__ En2,
                                               float* __restrict__ Out,
                                               float* __restrict__ Parts) {
  __shared__ __align__(16) unsigned short sE[2][CHUNK_U];   // double-buffered chunk
  __shared__ int   sIdx[WAVES][16];
  __shared__ float sRed[TPB];

  const int tid  = threadIdx.x;
  const int lane = tid & 31;
  const int wv   = tid >> 5;
  const int half = lane >> 4;      // which K-half of A/B this lane holds
  const int l16  = lane & 15;

  const int tile = blockIdx.x * WAVES + wv;       // 512 blocks * 8 waves = 4096 tiles
  const int n0   = tile * 16;
  const int b    = n0 >> 12;                      // n = b*4096 + h*64 + w
  const int h    = (n0 >> 6) & 63;
  const int w0   = n0 & 63;                       // multiple of 16
  const int xrow = b * (DD * HH * WW) + h * WW + w0 + l16;   // + d*4096

  // ---- build A fragments once: x-tile row M = l16, bf16 hi/lo split ----
  // contiguous K order (matches B): half 0 -> K 0..15, half 1 -> K 16..31
  v16bf Ah[2], Al[2];
  #pragma unroll
  for (int s = 0; s < 2; ++s) {
    us16 th, tl;
    #pragma unroll
    for (int p = 0; p < 16; ++p) {
      int kk = half * 16 + p;
      int d  = s * 32 + kk;
      float v = X[xrow + d * (HH * WW)];          // 2x 64B segments per wave: coalesced
      unsigned short hi = f2bf(v);
      th[p] = hi;
      tl[p] = f2bf(v - bf2f(hi));
    }
    Ah[s] = __builtin_bit_cast(v16bf, th);
    Al[s] = __builtin_bit_cast(v16bf, tl);
  }

  float minv[8];
  int   mini[8];
  #pragma unroll
  for (int j = 0; j < 8; ++j) { minv[j] = 3.4e38f; mini[j] = 0; }

  // ---- stage chunk 0 ----
#if HAVE_TDM
  if (wv == 0) {
    tdm_load_chunk(Epk, &sE[0][0]);
    __builtin_amdgcn_s_wait_tensorcnt(0);
  }
#else
  {
    const uint4* src = (const uint4*)(Epk);
    uint4* dst = (uint4*)(sE[0]);
    #pragma unroll
    for (int i = 0; i < 4; ++i) dst[tid + i * TPB] = src[tid + i * TPB];
  }
#endif
  __syncthreads();

  for (int ck = 0; ck < NCHUNK; ++ck) {
    const int buf = ck & 1;
#if HAVE_TDM
    if (wv == 0 && ck + 1 < NCHUNK)               // async-prefetch next chunk via TDM
      tdm_load_chunk(Epk + (ck + 1) * CHUNK_U, &sE[buf ^ 1][0]);
#else
    if (ck + 1 < NCHUNK) {
      const uint4* src = (const uint4*)(Epk + (ck + 1) * CHUNK_U);
      uint4* dst = (uint4*)(sE[buf ^ 1]);
      #pragma unroll
      for (int i = 0; i < 4; ++i) dst[tid + i * TPB] = src[tid + i * TPB];
    }
#endif
    const unsigned short* bh = &sE[buf][0];
    const unsigned short* bl = bh + CHUNK_H;
    #pragma unroll
    for (int g = 0; g < 4; ++g) {                 // 16 codes per WMMA group
      const int cl   = g * 16 + l16;              // local code column (B: N = lane&15)
      const int boff = cl * DD + half * 16;       // half 0: K 0..15, half 1: K 16..31
      v16bf Bh0 = *(const v16bf*)(bh + boff);
      v16bf Bh1 = *(const v16bf*)(bh + boff + 32);
      v16bf Bl0 = *(const v16bf*)(bl + boff);
      v16bf Bl1 = *(const v16bf*)(bl + boff + 32);
      v8f acc = {};
      // dot = xh*eh + xh*el + xl*eh  (f32 accumulate; error ~2^-15 relative)
      acc = __builtin_amdgcn_wmma_f32_16x16x32_bf16(false, Ah[0], false, Bh0, (short)0, acc, false, false);
      acc = __builtin_amdgcn_wmma_f32_16x16x32_bf16(false, Ah[1], false, Bh1, (short)0, acc, false, false);
      acc = __builtin_amdgcn_wmma_f32_16x16x32_bf16(false, Ah[0], false, Bl0, (short)0, acc, false, false);
      acc = __builtin_amdgcn_wmma_f32_16x16x32_bf16(false, Ah[1], false, Bl1, (short)0, acc, false, false);
      acc = __builtin_amdgcn_wmma_f32_16x16x32_bf16(false, Al[0], false, Bh0, (short)0, acc, false, false);
      acc = __builtin_amdgcn_wmma_f32_16x16x32_bf16(false, Al[1], false, Bh1, (short)0, acc, false, false);
      const int c  = ck * KC + cl;
      const float en = En2[c];
      #pragma unroll
      for (int j = 0; j < 8; ++j) {               // dist = ||e||^2 - 2 x.e  (||x||^2 const)
        float dv = en - 2.0f * acc[j];
        bool  lt = dv < minv[j];                  // strict <: first (smallest c) wins ties
        minv[j] = lt ? dv : minv[j];
        mini[j] = lt ? c  : mini[j];
      }
    }
#if HAVE_TDM
    if (wv == 0 && ck + 1 < NCHUNK)
      __builtin_amdgcn_s_wait_tensorcnt(0);       // next buffer resident before release
#endif
    __syncthreads();
  }

  // ---- 16-lane butterfly argmin within each half (rows j / j+8) ----
  #pragma unroll
  for (int j = 0; j < 8; ++j) {
    #pragma unroll
    for (int m = 1; m < 16; m <<= 1) {
      float ov = __shfl_xor(minv[j], m, 32);
      int   oi = __shfl_xor(mini[j], m, 32);
      if (ov < minv[j] || (ov == minv[j] && oi < mini[j])) { minv[j] = ov; mini[j] = oi; }
    }
    if (l16 == 0) sIdx[wv][j + 8 * half] = mini[j];
  }
  __syncthreads();

  // ---- write encoding indices (as float) ----
  if (lane < 16) Out[IDX_BASE + n0 + lane] = (float)sIdx[wv][lane];

  // ---- gather codebook (exact f32), write quantized, accumulate loss ----
  const int myIdx = sIdx[wv][l16];
  float lsum = 0.f;
  #pragma unroll 4
  for (int dd = 0; dd < 32; ++dd) {
    int d = 2 * dd + half;
    int off = b * (DD * HH * WW) + d * (HH * WW) + h * WW + w0 + l16;
    float q  = E[d * KK + myIdx];                 // L2-resident gather
    float xv = X[off];
    float df = xv - q;
    lsum += df * df;
    Out[off] = q;                                 // straight-through value == quantized
  }

  sRed[tid] = lsum;
  __syncthreads();
  for (int st = TPB / 2; st > 0; st >>= 1) {
    if (tid < st) sRed[tid] += sRed[tid + st];
    __syncthreads();
  }
  if (tid == 0) Parts[blockIdx.x] = sRed[0];
}

// ---------------- kernel 2: deterministic loss reduction ----------------
__global__ __launch_bounds__(256) void vq_loss(const float* __restrict__ Parts,
                                               float* __restrict__ Out) {
  __shared__ float s[256];
  float v = Parts[threadIdx.x] + Parts[threadIdx.x + 256];  // 512 block partials
  s[threadIdx.x] = v;
  __syncthreads();
  for (int st = 128; st > 0; st >>= 1) {
    if (threadIdx.x < st) s[threadIdx.x] += s[threadIdx.x + st];
    __syncthreads();
  }
  if (threadIdx.x == 0) {
    float m = s[0] * (1.0f / (float)QELEMS);
    Out[QELEMS]     = m;   // dictionary_loss
    Out[QELEMS + 1] = m;   // commitment_loss (numerically identical)
  }
}

// ---------------- host launcher ----------------
extern "C" void kernel_launch(void* const* d_in, const int* in_sizes, int n_in,
                              void* d_out, int out_size, void* d_ws, size_t ws_size,
                              hipStream_t stream) {
  (void)in_sizes; (void)n_in; (void)out_size; (void)ws_size;
  const float* X = (const float*)d_in[0];   // [16,64,64,64] f32
  const float* E = (const float*)d_in[1];   // [64,1024] f32
  float* Out = (float*)d_out;

  char* ws = (char*)d_ws;                   // ws layout (~268 KB):
  unsigned short* Epk = (unsigned short*)(ws);            // 256 KB packed bf16 chunks
  float* En2   = (float*)(ws + 262144);                   //   4 KB  ||e_k||^2
  float* Parts = (float*)(ws + 266240);                   //   2 KB  512 block partials

  vq_prep<<<dim3(4),   dim3(256), 0, stream>>>(E, Epk, En2);
  vq_main<<<dim3(512), dim3(TPB), 0, stream>>>(X, E, Epk, En2, Out, Parts);
  vq_loss<<<dim3(1),   dim3(256), 0, stream>>>(Parts, Out);
}
